// SAGE_31396210934185
// MI455X (gfx1250) — compile-verified, matrix operand order checked by reading, not run on hardware
//
#include <hip/hip_runtime.h>
#include <math.h>

typedef __attribute__((ext_vector_type(16))) _Float16 v16h;
typedef __attribute__((ext_vector_type(8)))  _Float16 v8h;
typedef __attribute__((ext_vector_type(4)))  _Float16 v4h;
typedef __attribute__((ext_vector_type(8)))  float    v8f;

#define DH 128

// ---------------------------------------------------------------- utilities

__global__ void zero_f32_kernel(float* __restrict__ p, long n) {
    long i = (long)blockIdx.x * blockDim.x + threadIdx.x;
    long stride = (long)gridDim.x * blockDim.x;
    for (; i < n; i += stride) p[i] = 0.0f;
}

__global__ void cvt_f32_to_f16_kernel(const float* __restrict__ in,
                                      _Float16* __restrict__ out, long n) {
    long i = (long)blockIdx.x * blockDim.x + threadIdx.x;
    if (i < n) out[i] = (_Float16)in[i];
}

// W is [128][128] row-major (k-major: W[k][n]); Wt[n][k] = W[k][n], f16
__global__ void transpose_w_kernel(const float* __restrict__ W,
                                   _Float16* __restrict__ Wt) {
    int n = blockIdx.x * 16 + threadIdx.x;
    int k = blockIdx.y * 16 + threadIdx.y;
    Wt[n * DH + k] = (_Float16)W[k * DH + n];
}

// ---------------------------------------------------------------- degree

__global__ void degree_kernel(const int* __restrict__ dst, float* __restrict__ cnt, int E) {
    int e = blockIdx.x * blockDim.x + threadIdx.x;
    if (e < E) atomicAdd(&cnt[dst[e]], 1.0f);
}

__global__ void inv_cnt_kernel(const float* __restrict__ cnt, float* __restrict__ inv, int n) {
    int i = blockIdx.x * blockDim.x + threadIdx.x;
    if (i < n) inv[i] = 1.0f / fmaxf(cnt[i], 1.0f);
}

// ---------------------------------------------------------------- scatter (mean agg numerator)
// gid -> (edge, 4-feature chunk); f16 gather (L2-resident), f32 atomic accumulate.

__global__ void scatter_kernel(const int* __restrict__ src, const int* __restrict__ dst,
                               const _Float16* __restrict__ feat,
                               float* __restrict__ agg, int E) {
    int gid = blockIdx.x * blockDim.x + threadIdx.x;
    int e = gid >> 5;
    if (e >= E) return;
    int c = (gid & 31) * 4;
    int s = src[e], d = dst[e];
    v4h v = *(const v4h*)(feat + (size_t)s * DH + c);
    float* a = agg + (size_t)d * DH + c;
    atomicAdd(a + 0, (float)v[0]);
    atomicAdd(a + 1, (float)v[1]);
    atomicAdd(a + 2, (float)v[2]);
    atomicAdd(a + 3, (float)v[3]);
}

// scale by 1/deg and narrow to f16 (WMMA A-matrix feed)
__global__ void agg_scale_cvt_kernel(const float* __restrict__ agg,
                                     const float* __restrict__ inv,
                                     _Float16* __restrict__ agg16, long n) {
    long i = (long)blockIdx.x * blockDim.x + threadIdx.x;
    if (i < n) agg16[i] = (_Float16)(agg[i] * inv[i >> 7]);
}

// ---------------------------------------------------------------- WMMA GEMM + norm epilogue
// 16-bit A/B 16x32 tile layout (ISA 7.12.2): lane = row (A) / col (B),
// lane[4] selects K-halves; halves 0..7 = K(k0 + hi8 .. +7), 8..15 = K(k0+16+hi8 ..).
__device__ inline v16h load_tile16(const _Float16* __restrict__ base, int k0, int lane) {
    int row = lane & 15;
    int hi8 = (lane >> 4) << 3;
    const _Float16* p = base + row * DH + k0 + hi8;
    v8h lo = *(const v8h*)(p);
    v8h hh = *(const v8h*)(p + 16);
    v16h r;
#pragma unroll
    for (int i = 0; i < 8; ++i) { r[i] = lo[i]; r[i + 8] = hh[i]; }
    return r;
}

// block = 256 threads = 8 wave32; block owns 16 rows, wave w owns cols [16w,16w+16)
// out = agg16 @ Wl^T' + bias + xin16 @ Wr^T'  ->  L2 normalize row  ->  (relu)  -> f16
__global__ __launch_bounds__(256) void sage_gemm_norm_kernel(
    const _Float16* __restrict__ agg16, const _Float16* __restrict__ xin16,
    const _Float16* __restrict__ Wlt,   const _Float16* __restrict__ Wrt,
    const float* __restrict__ bias, _Float16* __restrict__ hout, int do_relu) {
    __shared__ float tile[16][DH];
    __shared__ float invn[16];

    int tid  = threadIdx.x;
    int lane = tid & 31;
    int w    = tid >> 5;
    size_t rowblk = (size_t)blockIdx.x * 16;

    const _Float16* A1 = agg16 + rowblk * DH;
    const _Float16* A2 = xin16 + rowblk * DH;
    const _Float16* B1 = Wlt + (size_t)w * 16 * DH;
    const _Float16* B2 = Wrt + (size_t)w * 16 * DH;

    v8f acc = {};
#pragma unroll
    for (int k0 = 0; k0 < DH; k0 += 32) {
        v16h a1 = load_tile16(A1, k0, lane);
        v16h b1 = load_tile16(B1, k0, lane);
        acc = __builtin_amdgcn_wmma_f32_16x16x32_f16(false, a1, false, b1,
                                                     (short)0, acc, false, false);
        v16h a2 = load_tile16(A2, k0, lane);
        v16h b2 = load_tile16(B2, k0, lane);
        acc = __builtin_amdgcn_wmma_f32_16x16x32_f16(false, a2, false, b2,
                                                     (short)0, acc, false, false);
    }

    // D layout: vgpr v, lanes 0-15 -> M=v, lanes 16-31 -> M=8+v; N = lane&15
    int col = (lane & 15) + w * 16;
    int mhi = (lane >> 4) << 3;
    float bv = bias[col];
#pragma unroll
    for (int v = 0; v < 8; ++v) tile[mhi + v][col] = acc[v] + bv;
    __syncthreads();

    if (tid < 16) {
        float s = 0.0f;
#pragma unroll 4
        for (int c = 0; c < DH; ++c) { float t = tile[tid][c]; s += t * t; }
        invn[tid] = 1.0f / fmaxf(sqrtf(s), 1e-12f);
    }
    __syncthreads();

#pragma unroll
    for (int i = 0; i < 8; ++i) {
        int idx = tid + i * 256;           // 16*128 elements
        int r = idx >> 7, c = idx & 127;
        float vv = tile[r][c] * invn[r];
        if (do_relu) vv = fmaxf(vv, 0.0f);
        hout[(rowblk + (size_t)r) * DH + c] = (_Float16)vv;
    }
}

// ---------------------------------------------------------------- FC (128->40) + softmax

__global__ __launch_bounds__(128) void fc_softmax_kernel(
    const _Float16* __restrict__ h, const float* __restrict__ Wfc,
    const float* __restrict__ bfc, float* __restrict__ out, int N) {
    __shared__ float hv[DH];
    __shared__ float logits[40];
    __shared__ float red[2];
    int n = blockIdx.x;
    if (n >= N) return;
    int t = threadIdx.x;
    hv[t] = (float)h[(size_t)n * DH + t];
    __syncthreads();
    if (t < 40) {
        float s = bfc[t];
#pragma unroll 4
        for (int k = 0; k < DH; ++k) s += hv[k] * Wfc[k * 40 + t];
        logits[t] = s;
    }
    __syncthreads();
    if (t == 0) {
        float m = logits[0];
        for (int i = 1; i < 40; ++i) m = fmaxf(m, logits[i]);
        float ss = 0.0f;
        for (int i = 0; i < 40; ++i) ss += expf(logits[i] - m);
        red[0] = m; red[1] = 1.0f / ss;
    }
    __syncthreads();
    if (t < 40) out[(size_t)n * 40 + t] = expf(logits[t] - red[0]) * red[1];
}

// ---------------------------------------------------------------- driver

extern "C" void kernel_launch(void* const* d_in, const int* in_sizes, int n_in,
                              void* d_out, int out_size, void* d_ws, size_t ws_size,
                              hipStream_t stream) {
    const float* x   = (const float*)d_in[0];
    const int*   ei  = (const int*)d_in[1];
    const float* Wl[3] = { (const float*)d_in[2], (const float*)d_in[5], (const float*)d_in[8] };
    const float* bb[3] = { (const float*)d_in[3], (const float*)d_in[6], (const float*)d_in[9] };
    const float* Wr[3] = { (const float*)d_in[4], (const float*)d_in[7], (const float*)d_in[10] };
    const float* Wfc = (const float*)d_in[11];
    const float* bfc = (const float*)d_in[12];

    const int N  = in_sizes[0] / DH;
    const int E  = in_sizes[1] / 2;
    const int Np = (N + 15) & ~15;          // pad to 16 rows for WMMA tiling
    const long nf = (long)Np * DH;

    // workspace carve-up (256B aligned)
    char* ws = (char*)d_ws;
    size_t off = 0;
    auto carve = [&](size_t bytes) -> void* {
        void* p = (void*)(ws + off);
        off += (bytes + 255) & ~(size_t)255;
        return p;
    };
    _Float16* x16   = (_Float16*)carve(nf * 2);
    _Float16* h16a  = (_Float16*)carve(nf * 2);
    _Float16* h16b  = (_Float16*)carve(nf * 2);
    _Float16* agg16 = (_Float16*)carve(nf * 2);
    float*    agg32 = (float*)carve(nf * 4);
    float*    cnt   = (float*)carve((size_t)Np * 4);
    float*    inv   = (float*)carve((size_t)Np * 4);
    _Float16* wt    = (_Float16*)carve((size_t)6 * DH * DH * 2);
    (void)ws_size; (void)n_in;

    const int* srcI = ei;
    const int* dstI = ei + E;

    const int TB = 256;
    dim3 tw_grid(8, 8), tw_blk(16, 16);

    // one-time per call: degree, x->f16, weight transposes
    zero_f32_kernel<<<256, TB, 0, stream>>>(cnt, Np);
    degree_kernel<<<(E + TB - 1) / TB, TB, 0, stream>>>(dstI, cnt, E);
    inv_cnt_kernel<<<(Np + TB - 1) / TB, TB, 0, stream>>>(cnt, inv, Np);
    cvt_f32_to_f16_kernel<<<(nf + TB - 1) / TB, TB, 0, stream>>>(x, x16, (long)N * DH);
    for (int l = 0; l < 3; ++l) {
        transpose_w_kernel<<<tw_grid, tw_blk, 0, stream>>>(Wl[l], wt + (size_t)(2 * l) * DH * DH);
        transpose_w_kernel<<<tw_grid, tw_blk, 0, stream>>>(Wr[l], wt + (size_t)(2 * l + 1) * DH * DH);
    }

    const _Float16* fin[3]  = { x16,  h16a, h16b };
    _Float16*       fout[3] = { h16a, h16b, h16a };

    long sgrid = ((long)E * 32 + TB - 1) / TB;
    for (int l = 0; l < 3; ++l) {
        zero_f32_kernel<<<1024, TB, 0, stream>>>(agg32, nf);
        scatter_kernel<<<(int)sgrid, TB, 0, stream>>>(srcI, dstI, fin[l], agg32, E);
        agg_scale_cvt_kernel<<<(nf + TB - 1) / TB, TB, 0, stream>>>(agg32, inv, agg16, nf);
        sage_gemm_norm_kernel<<<Np / 16, TB, 0, stream>>>(
            agg16, fin[l],
            wt + (size_t)(2 * l) * DH * DH, wt + (size_t)(2 * l + 1) * DH * DH,
            bb[l], fout[l], (l < 2) ? 1 : 0);
    }

    fc_softmax_kernel<<<N, 128, 0, stream>>>(h16a, Wfc, bfc, (float*)d_out, N);
}
